// SoftMoE10_3719441678658
// MI455X (gfx1250) — matrix operand order repaired and checked by reading, not compiled, not run on hardware
//
#include <hip/hip_runtime.h>
#include <hip/hip_bf16.h>
#include <math.h>

// ---------------------------------------------------------------------------
// SoftMoE forward for MI455X (gfx1250, wave32, WMMA).
// Heavy GEMMs run in bf16 via v_wmma_f32_16x16x32_bf16 with f32 accumulation,
// double-buffered LDS staging; A tiles use GLOBAL_LOAD_ASYNC_TO_LDS_B128
// (ASYNCcnt). B-tile transpose ds_stores are issued AFTER the WMMA block so
// the matrix ops only wait on their own fragment ds_loads.
// ---------------------------------------------------------------------------

typedef __bf16 bf16;
typedef __attribute__((ext_vector_type(16))) __bf16 bf16x16;
typedef __attribute__((ext_vector_type(8)))  __bf16 bf16x8;
typedef __attribute__((ext_vector_type(8)))  float  f32x8;
typedef __attribute__((ext_vector_type(4)))  int    i32x4;

#define B_   8
#define N_   2048
#define D_   1024
#define E_   8
#define H_   512
#define S_   256
#define ES_  2048          // E_*S_
#define SEQ_ 2049

#if __has_builtin(__builtin_amdgcn_global_load_async_to_lds_b128)
#define ASYNC_LDS 1
#else
#define ASYNC_LDS 0
#endif

#if ASYNC_LDS
__device__ __forceinline__ void async_copy16(const void* g, void* l) {
  // builtin expects pointers to 16-byte vector data (int4*), generic AS
  __builtin_amdgcn_global_load_async_to_lds_b128((i32x4*)g, (i32x4*)l, 0, 0);
}
__device__ __forceinline__ void wait_async0() {
#if __has_builtin(__builtin_amdgcn_s_wait_asynccnt)
  __builtin_amdgcn_s_wait_asynccnt(0);
#else
  asm volatile("s_wait_asynccnt 0x0" ::: "memory");
#endif
}
#endif

__device__ __forceinline__ float gelu_exact(float x) {
  return 0.5f * x * (1.0f + erff(x * 0.70710678118654752f));
}

__device__ __forceinline__ float blockSum256(float v) {
  __shared__ float red[256];
  int t = threadIdx.x;
  red[t] = v; __syncthreads();
  #pragma unroll
  for (int s = 128; s > 0; s >>= 1) { if (t < s) red[t] += red[t + s]; __syncthreads(); }
  float r = red[0]; __syncthreads();
  return r;
}

__device__ __forceinline__ float blockMax256(float v) {
  __shared__ float red[256];
  int t = threadIdx.x;
  red[t] = v; __syncthreads();
  #pragma unroll
  for (int s = 128; s > 0; s >>= 1) { if (t < s) red[t] = fmaxf(red[t], red[t + s]); __syncthreads(); }
  float r = red[0]; __syncthreads();
  return r;
}

// ---------------------------------------------------------------------------
// Generic tiled bf16 GEMM: C(z) = A(z) * B(z%bMod), row-major everywhere.
// Block tile 128x128, K-step 32; 8 waves, wave tile 32x64 (2x4 WMMA frags).
// EPI: 0 = store f32, 1 = store bf16, 2 = +bias,GELU -> bf16, 3 = +bias -> bf16
// ---------------------------------------------------------------------------
template <int EPI>
__global__ __launch_bounds__(256)
void gemm_bf16_wmma(const bf16* __restrict__ A, long strideA, int lda,
                    const bf16* __restrict__ B, long strideB, int ldb, int bMod,
                    void* __restrict__ C, long strideC, int ldc,
                    const float* __restrict__ bias, int biasStride, int K) {
  constexpr int BM = 128, BN = 128, BK = 32;
  constexpr int LDSS = BK + 8;                         // 40 bf16 -> 80B rows
  __shared__ bf16 As[2][BM * LDSS];                    // [m][k]
  __shared__ bf16 Bst[2][BN * LDSS];                   // [n][k] (B transposed)

  const int tid  = threadIdx.x;
  const int lane = tid & 31;
  const int wave = tid >> 5;
  const int waveM = wave & 3;                          // 4 waves along M
  const int waveN = wave >> 2;                         // 2 waves along N
  const int hi = lane >> 4;
  const int lo = lane & 15;

  const long zA = (long)blockIdx.z * strideA;
  const long zB = (long)(blockIdx.z % bMod) * strideB;
  const long zC = (long)blockIdx.z * strideC;
  const int m0 = blockIdx.y * BM;
  const int n0 = blockIdx.x * BN;

  // per-thread staging coordinates (constant)
  const int a_r = tid >> 2;                // 0..63  (also +64 for 2nd chunk)
  const int a_c = (tid & 3) * 8;           // 0/8/16/24
  const int b_r = tid >> 4;                // 0..15  (also +16 for 2nd chunk)
  const int b_c = (tid & 15) * 8;          // 0..120

  const bf16* Ag = A + zA + (long)(m0 + a_r) * lda + a_c;   // + k0
  const bf16* Bg = B + zB + (long)b_r * ldb + (n0 + b_c);   // + k0*ldb
  const long  A64 = (long)64 * lda;
  const long  B16 = (long)16 * ldb;

  // issue the VMEM side of staging for tile at k0 (async A -> LDS, B -> regs)
  bf16x8 nv0, nv1;
  auto stage_issue = [&](int buf, int k0) {
    const bf16* ga = Ag + k0;
#if ASYNC_LDS
    async_copy16(ga,       &As[buf][a_r * LDSS + a_c]);
    async_copy16(ga + A64, &As[buf][(a_r + 64) * LDSS + a_c]);
#else
    *(bf16x8*)&As[buf][a_r * LDSS + a_c]        = *(const bf16x8*)ga;
    *(bf16x8*)&As[buf][(a_r + 64) * LDSS + a_c] = *(const bf16x8*)(ga + A64);
#endif
    const bf16* gb = Bg + (long)k0 * ldb;
    nv0 = *(const bf16x8*)gb;
    nv1 = *(const bf16x8*)(gb + B16);
    if (k0 + BK < K) __builtin_prefetch(gb + (long)BK * ldb, 0, 1);
  };
  // LDS-transpose store of the staged B data
  auto stage_store = [&](int buf) {
    #pragma unroll
    for (int j = 0; j < 8; ++j) {
      Bst[buf][(b_c + j) * LDSS + b_r]      = nv0[j];
      Bst[buf][(b_c + j) * LDSS + b_r + 16] = nv1[j];
    }
  };

  f32x8 acc[2][4];
  #pragma unroll
  for (int i = 0; i < 2; ++i)
    #pragma unroll
    for (int j = 0; j < 4; ++j) { f32x8 z = {}; acc[i][j] = z; }

  // prologue: fully stage tile 0
  stage_issue(0, 0);
  stage_store(0);
#if ASYNC_LDS
  wait_async0();
#endif

  const int nt = K / BK;
  for (int kt = 0; kt < nt; ++kt) {
    const int buf = kt & 1;
    const bool has_next = (kt + 1 < nt);
    __syncthreads();                       // tile kt (async A + ds B) visible

    // ---- fragments: contiguous k = 16*hi + e in both A and B ----
    bf16x16 afr[2], bfr[4];
    #pragma unroll
    for (int fm = 0; fm < 2; ++fm) {
      const bf16* pa = &As[buf][(waveM * 32 + fm * 16 + lo) * LDSS + 16 * hi];
      #pragma unroll
      for (int e = 0; e < 8; ++e) { afr[fm][e] = pa[e]; afr[fm][8 + e] = pa[8 + e]; }
    }
    #pragma unroll
    for (int fn = 0; fn < 4; ++fn) {
      const bf16* pb = &Bst[buf][(waveN * 64 + fn * 16 + lo) * LDSS + 16 * hi];
      #pragma unroll
      for (int e = 0; e < 16; ++e) bfr[fn][e] = pb[e];
    }

    // ---- VMEM staging of next tile goes in flight under the WMMAs ----
    if (has_next) stage_issue(buf ^ 1, (kt + 1) * BK);

    // ---- 8 WMMAs wait only on their own fragment ds_loads ----
    #pragma unroll
    for (int fm = 0; fm < 2; ++fm)
      #pragma unroll
      for (int fn = 0; fn < 4; ++fn)
        acc[fm][fn] = __builtin_amdgcn_wmma_f32_16x16x32_bf16(
            false, afr[fm], false, bfr[fn], (short)0, acc[fm][fn], false, false);

    // ---- transpose-store B for the next tile, then retire own asyncs ----
    if (has_next) {
      stage_store(buf ^ 1);
#if ASYNC_LDS
      wait_async0();
#endif
    }
  }

  // ---- epilogue ----
  const float* bptr = (EPI >= 2) ? (bias + (long)(blockIdx.z % bMod) * biasStride) : nullptr;
  #pragma unroll
  for (int fm = 0; fm < 2; ++fm) {
    #pragma unroll
    for (int fn = 0; fn < 4; ++fn) {
      int col = n0 + waveN * 64 + fn * 16 + lo;
      #pragma unroll
      for (int r = 0; r < 8; ++r) {
        int row = m0 + waveM * 32 + fm * 16 + (hi * 8 + r);
        float v = acc[fm][fn][r];
        if (EPI >= 2) v += bptr[col];
        if (EPI == 2) v = gelu_exact(v);
        long off = zC + (long)row * ldc + col;
        if (EPI == 0) ((float*)C)[off] = v;
        else          ((bf16*)C)[off]  = (bf16)v;
      }
    }
  }
}

// ---------------------------------------------------------------------------
// x_norm / xt -> bf16  (one block per token row)
// ---------------------------------------------------------------------------
__global__ __launch_bounds__(256)
void normalize_x_kernel(const float* __restrict__ x, bf16* __restrict__ xn,
                        bf16* __restrict__ xt) {
  int row = blockIdx.x;               // b*2048 + n
  int b = row >> 11, n = row & 2047;
  const float* rp = x + ((long)b * SEQ_ + 1 + n) * D_ + threadIdx.x * 4;
  float v[4]; float ss = 0.f;
  #pragma unroll
  for (int i = 0; i < 4; ++i) { v[i] = rp[i]; ss += v[i] * v[i]; }
  ss = blockSum256(ss);
  float inv = 1.0f / fmaxf(sqrtf(ss), 1e-12f);
  long o = (long)row * D_ + threadIdx.x * 4;
  #pragma unroll
  for (int i = 0; i < 4; ++i) { xn[o + i] = (bf16)(v[i] * inv); xt[o + i] = (bf16)v[i]; }
}

// mu column norm over D (+ scale) -> bf16, one block per es column
__global__ __launch_bounds__(256)
void normalize_mu_kernel(const float* __restrict__ mu, const float* __restrict__ scale,
                         bf16* __restrict__ mun) {
  int es = blockIdx.x, t = threadIdx.x;
  float v[4]; float ss = 0.f;
  #pragma unroll
  for (int i = 0; i < 4; ++i) { v[i] = mu[(long)(t + i * 256) * ES_ + es]; ss += v[i] * v[i]; }
  ss = blockSum256(ss);
  float inv = scale[0] / fmaxf(sqrtf(ss), 1e-12f);
  #pragma unroll
  for (int i = 0; i < 4; ++i) mun[(long)(t + i * 256) * ES_ + es] = (bf16)(v[i] * inv);
}

// transpose-convert (E,R,C) f32 -> (E,C,R) bf16
__global__ __launch_bounds__(256)
void transpose_convert_kernel(const float* __restrict__ in, bf16* __restrict__ out,
                              int R, int C) {
  long base = ((long)blockIdx.x * 256 + threadIdx.x) * 4;
  long rc = (long)R * C;
  #pragma unroll
  for (int i = 0; i < 4; ++i) {
    long id = base + i;
    long e = id / rc, rem = id % rc;
    long r = rem / C, c = rem % C;
    out[e * rc + c * R + r] = (bf16)in[id];
  }
}

// ---------------------------------------------------------------------------
// combine = softmax over es (row-wise), fused with m1 metric terms
// ---------------------------------------------------------------------------
__global__ __launch_bounds__(256)
void softmax_row_kernel(const float* __restrict__ logits, bf16* __restrict__ comb,
                        float* __restrict__ metrics) {
  __shared__ float s_diag;
  int row = blockIdx.x;                 // b*2048 + n
  int n_local = row & 2047;
  int t = threadIdx.x;
  const float* rp = logits + (long)row * ES_ + t * 8;
  float v[8]; float mx = -INFINITY;
  #pragma unroll
  for (int i = 0; i < 8; ++i) { v[i] = rp[i]; mx = fmaxf(mx, v[i]); }
  float smax = blockMax256(mx);
  float s = 0.f;
  #pragma unroll
  for (int i = 0; i < 8; ++i) { v[i] = expf(v[i] - smax); s += v[i]; }
  float S = blockSum256(s);
  float invS = 1.0f / S;
  float sq = 0.f;
  bf16* cp = comb + (long)row * ES_ + t * 8;
  #pragma unroll
  for (int i = 0; i < 8; ++i) { float p = v[i] * invS; sq += p * p; cp[i] = (bf16)p; }
  if (t == (n_local >> 3)) s_diag = v[n_local & 7] * invS;   // combine[b,n,es=n]
  float SS = blockSum256(sq);                                // syncs -> s_diag visible
  if (t == 0) {
    float r = rsqrtf(SS + 1e-9f);                            // row of tn sums to r
    atomicAdd(&metrics[0], r * r);
    atomicAdd(&metrics[1], s_diag * r * r);
  }
}

// dispatch = softmax over n (column-wise), writes transposed bf16 (b,es,n);
// fused with m2 metric terms (column sums of softmax are exactly 1).
__global__ __launch_bounds__(256)
void softmax_col_kernel(const float* __restrict__ logits, bf16* __restrict__ dispT,
                        float* __restrict__ metrics) {
  int b = blockIdx.x >> 11, es = blockIdx.x & 2047;
  int t = threadIdx.x;
  const float* base = logits + (long)b * N_ * ES_ + es;
  float v[8]; float mx = -INFINITY;
  #pragma unroll
  for (int j = 0; j < 8; ++j) { v[j] = base[(long)(t + j * 256) * ES_]; mx = fmaxf(mx, v[j]); }
  float smax = blockMax256(mx);
  float s = 0.f;
  #pragma unroll
  for (int j = 0; j < 8; ++j) { v[j] = expf(v[j] - smax); s += v[j]; }
  float S = blockSum256(s);
  float invS = 1.0f / S;
  float sq = 0.f;
  bf16* dp = dispT + ((long)b * ES_ + es) * N_;
  #pragma unroll
  for (int j = 0; j < 8; ++j) {
    float p = v[j] * invS; sq += p * p;
    dp[t + j * 256] = (bf16)p;
  }
  float SS = blockSum256(sq);
  if (t == 0) {
    float r = rsqrtf(SS + 1e-9f);
    atomicAdd(&metrics[2], r * r);
    if ((es % 257) == 0) atomicAdd(&metrics[3], r * r);      // (e=i,s=i) diagonal
  }
}

// ---------------------------------------------------------------------------
// cls-token MLP (8 rows, VALU; negligible work)
// ---------------------------------------------------------------------------
__global__ __launch_bounds__(256)
void cls_mlp_kernel(const float* __restrict__ x,
                    const float* __restrict__ w1, const float* __restrict__ b1,
                    const float* __restrict__ w2, const float* __restrict__ b2,
                    float* __restrict__ out) {
  __shared__ float xr[D_];
  __shared__ float hb[H_];
  int b = blockIdx.x, t = threadIdx.x;
  const float* xp = x + (long)b * SEQ_ * D_;
  for (int i = t; i < D_; i += 256) xr[i] = xp[i];
  __syncthreads();
  for (int h = t; h < H_; h += 256) {
    float a = b1[h];
    const float* wp = w1 + (long)h * D_;
    for (int d = 0; d < D_; ++d) a += wp[d] * xr[d];
    hb[h] = gelu_exact(a);
  }
  __syncthreads();
  float* op = out + (long)b * SEQ_ * D_;
  for (int d = t; d < D_; d += 256) {
    float a = b2[d];
    const float* wp = w2 + (long)d * H_;
    for (int h = 0; h < H_; ++h) a += wp[h] * hb[h];
    op[d] = a;
  }
}

__global__ void init_metrics_kernel(float* m) { if (threadIdx.x < 4) m[threadIdx.x] = 0.f; }

__global__ void finalize_metrics_kernel(const float* __restrict__ m, float* __restrict__ out) {
  out[0] = (m[0] - m[1]) / ((float)B_ * N_ * (N_ - 1));        // m1
  out[1] = (m[2] - m[3]) / ((float)B_ * E_ * S_ * (ES_ - 1));  // m2
}

// ---------------------------------------------------------------------------
extern "C" void kernel_launch(void* const* d_in, const int* in_sizes, int n_in,
                              void* d_out, int out_size, void* d_ws, size_t ws_size,
                              hipStream_t stream) {
  const float* x      = (const float*)d_in[0];
  const float* mu     = (const float*)d_in[1];
  const float* scale  = (const float*)d_in[2];
  const float* cls_w1 = (const float*)d_in[3];
  const float* cls_b1 = (const float*)d_in[4];
  const float* cls_w2 = (const float*)d_in[5];
  const float* cls_b2 = (const float*)d_in[6];
  const float* w1     = (const float*)d_in[7];
  const float* b1     = (const float*)d_in[8];
  const float* w2     = (const float*)d_in[9];
  const float* b2     = (const float*)d_in[10];
  float* out = (float*)d_out;

  char* ws = (char*)d_ws;
  // workspace layout (all 256B aligned)
  const long SZ_XN   = (long)B_ * N_ * D_ * 2;       // 33.55 MB
  const long SZ_MUN  = (long)D_ * ES_ * 2;           //  4.19 MB
  const long SZ_W1T  = (long)E_ * D_ * H_ * 2;       //  8.39 MB
  const long SZ_CMB  = (long)B_ * N_ * ES_ * 2;      // 67.11 MB
  long off = 256;
  float* MET = (float*)ws;
  bf16* XN   = (bf16*)(ws + off); off += SZ_XN;
  bf16* XT   = (bf16*)(ws + off); off += SZ_XN;
  bf16* MUN  = (bf16*)(ws + off); off += SZ_MUN;
  bf16* W1T  = (bf16*)(ws + off); off += SZ_W1T;
  bf16* W2T  = (bf16*)(ws + off); off += SZ_W1T;
  bf16* CMB  = (bf16*)(ws + off); off += SZ_CMB;
  bf16* DSPT = (bf16*)(ws + off); off += SZ_CMB;
  char* LOGB = ws + off;                             // logits f32; later reused:
  float* LOG = (float*)LOGB;
  bf16* SIN  = (bf16*)LOGB;                          // slot_in bf16 (aliases logits)
  bf16* HB   = (bf16*)(LOGB + SZ_XN);                // h bf16
  bf16* SOUT = (bf16*)(LOGB + SZ_XN + (long)B_ * E_ * S_ * H_ * 2);

  dim3 blk(256);

  init_metrics_kernel<<<1, 32, 0, stream>>>(MET);
  cls_mlp_kernel<<<B_, blk, 0, stream>>>(x, cls_w1, cls_b1, cls_w2, cls_b2, out);
  normalize_x_kernel<<<B_ * N_, blk, 0, stream>>>(x, XN, XT);
  normalize_mu_kernel<<<ES_, blk, 0, stream>>>(mu, scale, MUN);
  transpose_convert_kernel<<<4096, blk, 0, stream>>>(w1, W1T, H_, D_);  // -> (E,D,H)
  transpose_convert_kernel<<<4096, blk, 0, stream>>>(w2, W2T, D_, H_);  // -> (E,H,D)

  // logits[b] (2048x2048) = XN[b] (2048x1024) * MUN (1024x2048)
  gemm_bf16_wmma<0><<<dim3(ES_ / 128, N_ / 128, B_), blk, 0, stream>>>(
      XN, (long)N_ * D_, D_, MUN, 0, ES_, 1,
      LOG, (long)N_ * ES_, ES_, nullptr, 0, D_);

  softmax_row_kernel<<<B_ * N_, blk, 0, stream>>>(LOG, CMB, MET);
  softmax_col_kernel<<<B_ * ES_, blk, 0, stream>>>(LOG, DSPT, MET);

  // slot_in[b] (2048x1024) = dispatchT[b] (2048x2048) * xt[b] (2048x1024) -> bf16
  gemm_bf16_wmma<1><<<dim3(D_ / 128, ES_ / 128, B_), blk, 0, stream>>>(
      DSPT, (long)ES_ * N_, N_, XT, (long)N_ * D_, D_, B_,
      SIN, (long)ES_ * D_, D_, nullptr, 0, N_);

  // h[b,e] (256x512) = gelu(slot_in[b,e] (256x1024) * w1T[e] (1024x512) + b1[e])
  gemm_bf16_wmma<2><<<dim3(H_ / 128, S_ / 128, B_ * E_), blk, 0, stream>>>(
      SIN, (long)S_ * D_, D_, W1T, (long)D_ * H_, H_, E_,
      HB, (long)S_ * H_, H_, b1, H_, D_);

  // slot_out[b,e] (256x1024) = h[b,e] (256x512) * w2T[e] (512x1024) + b2[e]
  gemm_bf16_wmma<3><<<dim3(D_ / 128, S_ / 128, B_ * E_), blk, 0, stream>>>(
      HB, (long)S_ * H_, H_, W2T, (long)H_ * D_, D_, E_,
      SOUT, (long)S_ * D_, D_, b2, D_, H_);

  // img[b] (2048x1024) = combine[b] (2048x2048) * slot_out[b] (2048x1024) -> f32 out rows 1..2048
  gemm_bf16_wmma<0><<<dim3(D_ / 128, N_ / 128, B_), blk, 0, stream>>>(
      CMB, (long)N_ * ES_, ES_, SOUT, (long)ES_ * D_, D_, B_,
      out + D_, (long)SEQ_ * D_, D_, nullptr, 0, ES_);

  finalize_metrics_kernel<<<1, 1, 0, stream>>>(MET, out + (long)B_ * SEQ_ * D_);
}
